// LSTM_20177756356879
// MI455X (gfx1250) — compile-verified, matrix operand order checked by reading, not run on hardware
//
#include <hip/hip_runtime.h>
#include <stdint.h>

// ---------------- problem constants ----------------
#define BB   64
#define TT   2048
#define DD   64
#define HH   512
#define NG   2048      // 4*HH gate rows
#define NOUT 10

// persistent kernel geometry: 128 waves = 4 batch-tiles x 32 hidden-tiles
#define NBLK 16
#define NTHR 256

typedef __attribute__((ext_vector_type(16))) __bf16 v16bf;
typedef __attribute__((ext_vector_type(8)))  float  v8f;

union Frag { v16bf v; uint4 q[2]; };

__device__ __forceinline__ unsigned short f2bf(float x) {
  unsigned u = __float_as_uint(x);
  u += 0x7FFFu + ((u >> 16) & 1u);          // round-to-nearest-even
  return (unsigned short)(u >> 16);
}

__device__ __forceinline__ float sigm(float x) { return 1.0f / (1.0f + __expf(-x)); }

// tanh via the hardware exp trans op: tanh(x) = 1 - 2/(1+exp(2x)).
// exp(2x)->inf gives 1, exp(2x)->0 gives -1, so saturation is correct.
__device__ __forceinline__ float tanh_fast(float x) {
  return 1.0f - 2.0f / (1.0f + __expf(2.0f * x));
}

// One K-step of the 4-gate GEMM: load A fragment (batch rows) and the four
// gate B fragments (weight rows), then 4 WMMAs. All loads are SGPR-base +
// 32-bit VGPR element offset; the per-gate displacement g*HH*BSTRIDE and the
// +16 intra-fragment displacement are compile-time constants folded into the
// 24-bit instruction offset (max 3*512*512*2B = 1.5 MB < 8 MB).
template <int BSTRIDE>
__device__ __forceinline__ void mac4(v8f& d0, v8f& d1, v8f& d2, v8f& d3,
                                     const unsigned short* __restrict__ ab, int ao,
                                     const unsigned short* __restrict__ bb, int bo) {
  Frag a, b0, b1, b2, b3;
  a.q[0]  = *(const uint4*)(ab + ao);
  a.q[1]  = *(const uint4*)(ab + ao + 16);
  b0.q[0] = *(const uint4*)(bb + bo);
  b0.q[1] = *(const uint4*)(bb + bo + 16);
  b1.q[0] = *(const uint4*)(bb + bo + 1 * HH * BSTRIDE);
  b1.q[1] = *(const uint4*)(bb + bo + 1 * HH * BSTRIDE + 16);
  b2.q[0] = *(const uint4*)(bb + bo + 2 * HH * BSTRIDE);
  b2.q[1] = *(const uint4*)(bb + bo + 2 * HH * BSTRIDE + 16);
  b3.q[0] = *(const uint4*)(bb + bo + 3 * HH * BSTRIDE);
  b3.q[1] = *(const uint4*)(bb + bo + 3 * HH * BSTRIDE + 16);
  d0 = __builtin_amdgcn_wmma_f32_16x16x32_bf16(false, a.v, false, b0.v, (short)0, d0, false, false);
  d1 = __builtin_amdgcn_wmma_f32_16x16x32_bf16(false, a.v, false, b1.v, (short)0, d1, false, false);
  d2 = __builtin_amdgcn_wmma_f32_16x16x32_bf16(false, a.v, false, b2.v, (short)0, d2, false, false);
  d3 = __builtin_amdgcn_wmma_f32_16x16x32_bf16(false, a.v, false, b3.v, (short)0, d3, false, false);
}

// grid-wide split barrier (16 co-resident blocks); bar[0]=count, bar[1]=generation
__device__ __forceinline__ void gsync(int* cnt, volatile int* gen) {
  __threadfence();
  __syncthreads();
  if (threadIdx.x == 0) {
    int g = *gen;
    if (atomicAdd(cnt, 1) == NBLK - 1) {
      *cnt = 0;
      __threadfence();
      atomicAdd((int*)gen, 1);
    } else {
      while (*gen == g) __builtin_amdgcn_s_sleep(2);
    }
  }
  __syncthreads();
}

// ---------------- prep kernels ----------------
__global__ void k_init(int* __restrict__ bar,
                       unsigned short* __restrict__ h0buf,
                       unsigned short* __restrict__ h1buf) {
  int i = blockIdx.x * blockDim.x + threadIdx.x;
  if (i == 0) { bar[0] = 0; bar[1] = 0; }
  const int nh = 2 * BB * HH;
  for (int j = i; j < nh; j += gridDim.x * blockDim.x) { h0buf[j] = 0; h1buf[j] = 0; }
}

__global__ void k_cvt_bf16(const float* __restrict__ src,
                           unsigned short* __restrict__ dst, int n) {
  for (int i = blockIdx.x * blockDim.x + threadIdx.x; i < n; i += gridDim.x * blockDim.x)
    dst[i] = f2bf(src[i]);
}

__global__ void k_bias_sum(const float* __restrict__ a, const float* __restrict__ b,
                           float* __restrict__ o, int n) {
  int i = blockIdx.x * blockDim.x + threadIdx.x;
  if (i < n) o[i] = a[i] + b[i];
}

// ---------------- fused 2-layer persistent LSTM ----------------
__global__ void __launch_bounds__(NTHR, 1) k_lstm_persistent(
    const unsigned short* __restrict__ xbf,    // (B,T,D) bf16
    const unsigned short* __restrict__ wih0,   // (4H,D)  bf16
    const unsigned short* __restrict__ whh0,   // (4H,H)  bf16
    const unsigned short* __restrict__ wih1,   // (4H,H)  bf16
    const unsigned short* __restrict__ whh1,   // (4H,H)  bf16
    const float* __restrict__ bias0,           // (4H) b_ih0+b_hh0
    const float* __restrict__ bias1,
    unsigned short* __restrict__ h0buf,        // 2 x (B,H) bf16, parity double-buffer
    unsigned short* __restrict__ h1buf,
    float* __restrict__ h1last,                // (B,H) fp32, layer-1 h at t=T-1
    int* __restrict__ bar)
{
  const int lane = threadIdx.x & 31;
  const int hlf  = lane >> 4;                  // half-wave selects K-chunk pair
  const int col  = lane & 15;
  const int w    = (blockIdx.x << 3) + (threadIdx.x >> 5);   // wave id 0..127
  const int bt   = w >> 5;                     // batch tile 0..3
  const int ht   = w & 31;                     // hidden tile 0..31

  const int am   = bt * 16 + col;              // batch row this lane feeds into A
  const int ncol = ht * 16 + col;              // hidden unit column this lane owns

  // lane-constant 32-bit element offsets (SGPR base + VGPR offset addressing)
  const int lo8    = 8 * hlf;
  const int axBase = am * TT * DD + lo8;       // into xbf, add t*DD per step
  const int ahOff  = am * HH + lo8;            // into h buffers
  const int bD     = ncol * DD + lo8;          // into wih0 (gate 0 row)
  const int bH     = ncol * HH + lo8;          // into whh0/wih1/whh1 (gate 0 row)

  float b0v[4], b1v[4];
#pragma unroll
  for (int g = 0; g < 4; ++g) {
    b0v[g] = bias0[g * HH + ncol];
    b1v[g] = bias1[g * HH + ncol];
  }

  // cell states live in VGPRs for the whole sequence
  v8f c0, c1;
#pragma unroll
  for (int r = 0; r < 8; ++r) { c0[r] = 0.0f; c1[r] = 0.0f; }

  int* cnt = bar;
  volatile int* gen = (volatile int*)(bar + 1);

  for (int t = 0; t < TT; ++t) {
    const int p = t & 1;
    const unsigned short* h0_rd = h0buf + (size_t)(p ^ 1) * (BB * HH);
    unsigned short*       h0_wr = h0buf + (size_t)p       * (BB * HH);
    const unsigned short* h1_rd = h1buf + (size_t)(p ^ 1) * (BB * HH);
    unsigned short*       h1_wr = h1buf + (size_t)p       * (BB * HH);

    // ======== layer 0: gates = x_t @ W_ih0^T + h0 @ W_hh0^T + bias ========
    v8f a0, a1, a2, a3;
#pragma unroll
    for (int r = 0; r < 8; ++r) { a0[r] = b0v[0]; a1[r] = b0v[1]; a2[r] = b0v[2]; a3[r] = b0v[3]; }

    {
      const int ax = axBase + t * DD;
      mac4<DD>(a0, a1, a2, a3, xbf, ax,      wih0, bD);
      mac4<DD>(a0, a1, a2, a3, xbf, ax + 32, wih0, bD + 32);
#pragma unroll 2
      for (int k0 = 0; k0 < HH; k0 += 32)
        mac4<HH>(a0, a1, a2, a3, h0_rd, ahOff + k0, whh0, bH + k0);
    }

    // elementwise gate math; C layout: VGPR r, lane half -> M = r + 8*hlf
#pragma unroll
    for (int r = 0; r < 8; ++r) {
      const float ig = sigm(a0[r]);
      const float fg = sigm(a1[r]);
      const float gg = tanh_fast(a2[r]);
      const float og = sigm(a3[r]);
      const float c  = fg * c0[r] + ig * gg;
      c0[r] = c;
      const float h  = og * tanh_fast(c);
      const int row  = bt * 16 + r + 8 * hlf;
      h0_wr[(size_t)row * HH + ncol] = f2bf(h);
    }

    gsync(cnt, gen);   // single sync per step (double buffering covers the rest)

    // ======== layer 1: gates = h0_t @ W_ih1^T + h1 @ W_hh1^T + bias ========
#pragma unroll
    for (int r = 0; r < 8; ++r) { a0[r] = b1v[0]; a1[r] = b1v[1]; a2[r] = b1v[2]; a3[r] = b1v[3]; }

    {
#pragma unroll 2
      for (int k0 = 0; k0 < HH; k0 += 32)
        mac4<HH>(a0, a1, a2, a3, h0_wr, ahOff + k0, wih1, bH + k0);
#pragma unroll 2
      for (int k0 = 0; k0 < HH; k0 += 32)
        mac4<HH>(a0, a1, a2, a3, h1_rd, ahOff + k0, whh1, bH + k0);
    }

#pragma unroll
    for (int r = 0; r < 8; ++r) {
      const float ig = sigm(a0[r]);
      const float fg = sigm(a1[r]);
      const float gg = tanh_fast(a2[r]);
      const float og = sigm(a3[r]);
      const float c  = fg * c1[r] + ig * gg;
      c1[r] = c;
      const float h  = og * tanh_fast(c);
      const int row  = bt * 16 + r + 8 * hlf;
      h1_wr[(size_t)row * HH + ncol] = f2bf(h);
      if (t == TT - 1) h1last[(size_t)row * HH + ncol] = h;
    }
    // no end-of-step sync needed: next L0 touches only h0 buffers (disjoint
    // parity) and the mid-step sync of t+1 orders everything else.
  }
}

// ---------------- final linear + log_softmax (64 x 10) ----------------
__global__ void k_logits(const float* __restrict__ hlast,   // (B,H) fp32
                         const float* __restrict__ Wl,      // (10,H)
                         const float* __restrict__ bl,      // (10)
                         float* __restrict__ out) {         // (B,10)
  const int b = threadIdx.x;
  if (b >= BB) return;
  float lg[NOUT];
  const float* hr = hlast + (size_t)b * HH;
#pragma unroll
  for (int o = 0; o < NOUT; ++o) {
    float s = bl[o];
    const float* wr = Wl + (size_t)o * HH;
    for (int k = 0; k < HH; ++k) s += hr[k] * wr[k];
    lg[o] = s;
  }
  float m = lg[0];
#pragma unroll
  for (int o = 1; o < NOUT; ++o) m = fmaxf(m, lg[o]);
  float se = 0.0f;
#pragma unroll
  for (int o = 0; o < NOUT; ++o) se += __expf(lg[o] - m);
  const float lse = m + __logf(se);
#pragma unroll
  for (int o = 0; o < NOUT; ++o) out[b * NOUT + o] = lg[o] - lse;
}

extern "C" void kernel_launch(void* const* d_in, const int* in_sizes, int n_in,
                              void* d_out, int out_size, void* d_ws, size_t ws_size,
                              hipStream_t stream) {
  (void)in_sizes; (void)n_in; (void)out_size; (void)ws_size;
  const float* x     = (const float*)d_in[0];
  const float* W_ih0 = (const float*)d_in[1];
  const float* W_hh0 = (const float*)d_in[2];
  const float* b_ih0 = (const float*)d_in[3];
  const float* b_hh0 = (const float*)d_in[4];
  const float* W_ih1 = (const float*)d_in[5];
  const float* W_hh1 = (const float*)d_in[6];
  const float* b_ih1 = (const float*)d_in[7];
  const float* b_hh1 = (const float*)d_in[8];
  const float* W_lin = (const float*)d_in[9];
  const float* b_lin = (const float*)d_in[10];

  char* w = (char*)d_ws;
  size_t off = 0;
  auto alloc = [&](size_t bytes) -> char* {
    char* p = w + off;
    off += (bytes + 255) & ~(size_t)255;      // keep every region 256B-aligned
    return p;
  };
  int*            bar    = (int*)alloc(256);
  float*          bias0  = (float*)alloc((size_t)NG * 4);
  float*          bias1  = (float*)alloc((size_t)NG * 4);
  unsigned short* wih0   = (unsigned short*)alloc((size_t)NG * DD * 2);
  unsigned short* whh0   = (unsigned short*)alloc((size_t)NG * HH * 2);
  unsigned short* wih1   = (unsigned short*)alloc((size_t)NG * HH * 2);
  unsigned short* whh1   = (unsigned short*)alloc((size_t)NG * HH * 2);
  unsigned short* xbf    = (unsigned short*)alloc((size_t)BB * TT * DD * 2);
  unsigned short* h0buf  = (unsigned short*)alloc((size_t)2 * BB * HH * 2);
  unsigned short* h1buf  = (unsigned short*)alloc((size_t)2 * BB * HH * 2);
  float*          h1last = (float*)alloc((size_t)BB * HH * 4);

  // prep: reset barrier + zero h state, bf16-convert weights/x, fold biases
  k_init<<<64, 256, 0, stream>>>(bar, h0buf, h1buf);
  k_cvt_bf16<<<256, 256, 0, stream>>>(W_ih0, wih0, NG * DD);
  k_cvt_bf16<<<1024, 256, 0, stream>>>(W_hh0, whh0, NG * HH);
  k_cvt_bf16<<<1024, 256, 0, stream>>>(W_ih1, wih1, NG * HH);
  k_cvt_bf16<<<1024, 256, 0, stream>>>(W_hh1, whh1, NG * HH);
  k_cvt_bf16<<<4096, 256, 0, stream>>>(x, xbf, BB * TT * DD);
  k_bias_sum<<<(NG + 255) / 256, 256, 0, stream>>>(b_ih0, b_hh0, bias0, NG);
  k_bias_sum<<<(NG + 255) / 256, 256, 0, stream>>>(b_ih1, b_hh1, bias1, NG);

  // fused two-layer persistent recurrence (one grid sync per timestep)
  k_lstm_persistent<<<NBLK, NTHR, 0, stream>>>(xbf, wih0, whh0, wih1, whh1,
                                               bias0, bias1, h0buf, h1buf,
                                               h1last, bar);

  k_logits<<<1, 64, 0, stream>>>(h1last, W_lin, b_lin, (float*)d_out);
}
